// BispectrumPool_88227218195138
// MI455X (gfx1250) — matrix operand order repaired and checked by reading, not compiled, not run on hardware
//
#include <hip/hip_runtime.h>
#include <stdint.h>

// ---- types ---------------------------------------------------------------
typedef _Float16 half8 __attribute__((ext_vector_type(8)));
typedef _Float16 v16h  __attribute__((ext_vector_type(16)));
typedef float    v8f   __attribute__((ext_vector_type(8)));
typedef float    v4f   __attribute__((ext_vector_type(4)));

// ---- problem constants ---------------------------------------------------
#define BATCH   16
#define C_CH    64
#define G_AX    8
#define HW      3136          // 56*56
#define KDIM    1024          // C * 2G features
#define TILE_M  32            // pixels per block (3136 % 32 == 0)
#define THREADS 256           // 8 waves of 32
#define LDSTR   (KDIM + 8)    // padded row stride in halves (bank stagger)
#define LN2F    0.69314718055994530942f

// ---- prologue: conv_w f32 -> f16 into workspace (done once per launch) ----
__global__ void cvt_weights_kernel(const float* __restrict__ w,
                                   _Float16* __restrict__ wh) {
    const int i = (blockIdx.x * 256 + threadIdx.x) * 8;
    const v4f a = *(const v4f*)(w + i);
    const v4f b = *(const v4f*)(w + i + 4);
    half8 h;
    h[0] = (_Float16)a[0]; h[1] = (_Float16)a[1];
    h[2] = (_Float16)a[2]; h[3] = (_Float16)a[3];
    h[4] = (_Float16)b[0]; h[5] = (_Float16)b[1];
    h[6] = (_Float16)b[2]; h[7] = (_Float16)b[3];
    *(half8*)(wh + i) = h;
}

// ---- fused bispectrum + WMMA GEMM kernel ---------------------------------
template <int USE_ASYNC_W>
__global__ __launch_bounds__(THREADS, 1)
void bispectrum_wmma_kernel(const float* __restrict__ x,
                            const float* __restrict__ conv_w,
                            const _Float16* __restrict__ conv_wh,
                            const float* __restrict__ conv_b,
                            float* __restrict__ y)
{
    // A tile: 32 pixels x 1024 features (f16)   ~66 KB
    // B tile: 64 out-ch x 1024 features (f16)   ~132 KB  (B[k][n] == W[n][k])
    __shared__ __align__(16) _Float16 Alds[TILE_M * LDSTR];
    __shared__ __align__(16) _Float16 Wlds[C_CH  * LDSTR];

    const int t    = threadIdx.x;
    const int blk  = blockIdx.x;
    const int bIdx = blk / (HW / TILE_M);
    const int p0   = (blk % (HW / TILE_M)) * TILE_M;

    // ---- stage weights into LDS --------------------------------------------
    if (USE_ASYNC_W) {
        // CDNA5 async copy global(f16) -> LDS, overlapped with the feature phase.
        // LDS byte address = low 32 bits of the generic pointer (ISA: addr[31:0]).
        const uint32_t wbase = (uint32_t)(uintptr_t)&Wlds[0];
        const uint64_t gbase = (uint64_t)(uintptr_t)conv_wh;
        #pragma unroll
        for (int it = 0; it < (C_CH * KDIM) / (THREADS * 8); ++it) {   // 32 chunks
            const int idx = it * THREADS + t;          // 8-half chunk id
            const int o   = idx >> 7;                  // out channel
            const int kb  = (idx & 127) << 4;          // byte offset in row
            const uint32_t lds_addr = wbase + o * (LDSTR * 2) + kb;
            const uint64_t gaddr    = gbase + (uint64_t)((o << 11) + kb);
            asm volatile("global_load_async_to_lds_b128 %0, %1, off"
                         :: "v"(lds_addr), "v"(gaddr)
                         : "memory");
        }
    } else {
        // fallback: in-kernel convert f32 -> f16 (vectorized)
        #pragma unroll 4
        for (int it = 0; it < (C_CH * KDIM) / (THREADS * 8); ++it) {
            const int idx = it * THREADS + t;
            const int o   = idx >> 7;
            const int k   = (idx & 127) << 3;
            const float* src = conv_w + (o << 10) + k;
            const v4f a = *(const v4f*)(src);
            const v4f b = *(const v4f*)(src + 4);
            half8 h;
            h[0] = (_Float16)a[0]; h[1] = (_Float16)a[1];
            h[2] = (_Float16)a[2]; h[3] = (_Float16)a[3];
            h[4] = (_Float16)b[0]; h[5] = (_Float16)b[1];
            h[6] = (_Float16)b[2]; h[7] = (_Float16)b[3];
            *(half8*)&Wlds[o * LDSTR + k] = h;
        }
    }

    // ---- feature phase: 8-pt real FFT -> selective bispectrum -> relu(log1p) ----
    const float S = 0.70710678118654752440f;   // sqrt(2)/2
    #pragma unroll
    for (int it = 0; it < (TILE_M / 4 * C_CH) / THREADS; ++it) {   // 2 tasks/thread
        const int task = it * THREADS + t;
        const int grp  = task & 7;     // pixel quad (4 consecutive pixels)
        const int c    = task >> 3;    // channel
        const float* xb = x + ((size_t)(bIdx * C_CH + c) * G_AX) * HW + (p0 + grp * 4);

        v4f g[G_AX];
        #pragma unroll
        for (int gg = 0; gg < G_AX; ++gg)
            g[gg] = __builtin_nontemporal_load((const v4f*)(xb + gg * HW));

        #pragma unroll
        for (int q = 0; q < 4; ++q) {
            const float x0 = g[0][q], x1 = g[1][q], x2 = g[2][q], x3 = g[3][q];
            const float x4 = g[4][q], x5 = g[5][q], x6 = g[6][q], x7 = g[7][q];

            const float a0 = x0 + x4, a1 = x1 + x5, a2 = x2 + x6, a3 = x3 + x7;
            const float b0 = x0 - x4, b1 = x1 - x5, b2 = x2 - x6, b3 = x3 - x7;

            float Fr[8], Fi[8];
            Fr[0] = a0 + a1 + a2 + a3;  Fi[0] = 0.0f;
            Fr[4] = a0 - a1 + a2 - a3;  Fi[4] = 0.0f;
            Fr[2] = a0 - a2;            Fi[2] = a3 - a1;
            Fr[6] = Fr[2];              Fi[6] = -Fi[2];
            const float sd = S * (b1 - b3), ss = S * (b1 + b3);
            Fr[1] = b0 + sd;            Fi[1] = -(b2 + ss);
            Fr[7] = Fr[1];              Fi[7] = -Fi[1];
            Fr[3] = b0 - sd;            Fi[3] = b2 - ss;
            Fr[5] = Fr[3];              Fi[5] = -Fi[3];

            // beta[k] = F1 * F[k] * conj(F[(k+1)%8])
            // relu(signed_log1p(v)) == log1p(max(v,0)) == ln2 * log2(1 + max(v,0))
            half8 lo, hi;
            #pragma unroll
            for (int k = 0; k < 8; ++k) {
                const int j = (k + 1) & 7;
                const float tr = Fr[1] * Fr[k] - Fi[1] * Fi[k];
                const float ti = Fr[1] * Fi[k] + Fi[1] * Fr[k];
                const float ur = tr * Fr[j] + ti * Fi[j];
                const float ui = ti * Fr[j] - tr * Fi[j];
                lo[k] = (_Float16)(LN2F * __log2f(1.0f + fmaxf(ur, 0.0f)));
                hi[k] = (_Float16)(LN2F * __log2f(1.0f + fmaxf(ui, 0.0f)));
            }
            half8* dst = (half8*)&Alds[(grp * 4 + q) * LDSTR + c * 16];
            dst[0] = lo;   // real parts  (feat f = 0..7)
            dst[1] = hi;   // imag parts  (feat f = 8..15)
        }
    }

    if (USE_ASYNC_W)
        asm volatile("s_wait_asynccnt 0x0" ::: "memory");
    __syncthreads();

    // ---- GEMM phase: each wave computes one 16x16 tile of D = A(32x1024) * B(1024x64) ----
    const int wave  = t >> 5;
    const int lane  = t & 31;
    const int m0    = (wave >> 2) * 16;     // 0 or 16
    const int n0    = (wave & 3) * 16;      // 0,16,32,48
    const int laneM = lane & 15;
    const int hiH   = lane >> 4;            // 0: lanes 0-15, 1: lanes 16-31

    const _Float16* Arow = &Alds[(m0 + laneM) * LDSTR];   // A row = pixel
    const _Float16* Brow = &Wlds[(n0 + laneM) * LDSTR];   // B "row" = out channel
    const int abase = hiH * 16;   // A: lane<16 -> K 0..15; lane>=16 -> K 16..31
    const int bbase = hiH * 16;   // B: lane<16 -> K 0..15; lane>=16 -> K 16..31

    v8f acc = {};
    #pragma unroll 4
    for (int kk = 0; kk < KDIM; kk += 32) {
        const half8 alo = *(const half8*)(Arow + kk + abase);
        const half8 ahi = *(const half8*)(Arow + kk + abase + 8);
        const half8 blo = *(const half8*)(Brow + kk + bbase);
        const half8 bhi = *(const half8*)(Brow + kk + bbase + 8);
        const v16h a  = __builtin_shufflevector(alo, ahi,
                          0, 1, 2, 3, 4, 5, 6, 7, 8, 9, 10, 11, 12, 13, 14, 15);
        const v16h bm = __builtin_shufflevector(blo, bhi,
                          0, 1, 2, 3, 4, 5, 6, 7, 8, 9, 10, 11, 12, 13, 14, 15);
        acc = __builtin_amdgcn_wmma_f32_16x16x32_f16(
                  /*neg_a=*/false, a, /*neg_b=*/false, bm,
                  /*c_mod=*/(short)0, acc, /*reuse_a=*/false, /*reuse_b=*/false);
    }

    // ---- epilogue: bias + store. D layout: lane -> n = n0+laneM, vgpr r -> m = m0+r+8*hiH.
    const int   o    = n0 + laneM;
    const float bias = conv_b[o];
    float* yp = y + ((size_t)(bIdx * C_CH + o)) * HW + (p0 + m0 + hiH * 8);

    v4f s0 = { acc[0] + bias, acc[1] + bias, acc[2] + bias, acc[3] + bias };
    v4f s1 = { acc[4] + bias, acc[5] + bias, acc[6] + bias, acc[7] + bias };
    __builtin_nontemporal_store(s0, (v4f*)(yp + 0));
    __builtin_nontemporal_store(s1, (v4f*)(yp + 4));
}

extern "C" void kernel_launch(void* const* d_in, const int* in_sizes, int n_in,
                              void* d_out, int out_size, void* d_ws, size_t ws_size,
                              hipStream_t stream) {
    const float* x      = (const float*)d_in[0];   // (16,64,8,56,56) f32
    const float* conv_w = (const float*)d_in[1];   // (64,1024) f32
    const float* conv_b = (const float*)d_in[2];   // (64,) f32
    float*       y      = (float*)d_out;           // (16,64,56,56) f32

    const int blocks = (BATCH * HW) / TILE_M;      // 1568
    const size_t w_bytes = (size_t)C_CH * KDIM * sizeof(_Float16);   // 128 KB

    if (ws_size >= w_bytes) {
        _Float16* wh = (_Float16*)d_ws;
        cvt_weights_kernel<<<dim3((C_CH * KDIM) / (256 * 8)), dim3(256), 0, stream>>>(conv_w, wh);
        bispectrum_wmma_kernel<1><<<dim3(blocks), dim3(THREADS), 0, stream>>>(
            x, conv_w, wh, conv_b, y);
    } else {
        bispectrum_wmma_kernel<0><<<dim3(blocks), dim3(THREADS), 0, stream>>>(
            x, conv_w, (const _Float16*)nullptr, conv_b, y);
    }
}
